// JointModel_4492535792044
// MI455X (gfx1250) — matrix operand order repaired and checked by reading, not compiled, not run on hardware
//
#include <hip/hip_runtime.h>
#include <hip/hip_bf16.h>
#include <math.h>

typedef __bf16 bf16;
typedef __attribute__((ext_vector_type(16))) __bf16 v16bf;
typedef __attribute__((ext_vector_type(8)))  __bf16 v8bf;
typedef __attribute__((ext_vector_type(8)))  float  v8f;

#define B_   4
#define L_   4
#define SV   512
#define SP   1
#define SA   64
#define ST   577          // 512 + 1 + 64
#define D_   1024
#define NH   8
#define HD   128
#define QD   (NH*HD)      // 1024
#define KD   128
#define DFF  4096
#define QTILES 37         // ceil(577/16)
#define INV_SQRT_HD 0.08838834764831845f
#define SOFTCAP 50.0f

#define MODE_PLAIN 0
#define MODE_RESID 1
#define MODE_GEGLU 2

static __device__ inline v8f zero8() {
    v8f z;
#pragma unroll
    for (int i = 0; i < 8; ++i) z[i] = 0.0f;
    return z;
}

static __device__ inline float gelu_tanh(float x) {
    return 0.5f * x * (1.0f + tanhf(0.7978845608028654f * (x + 0.044715f * x * x * x)));
}

// ---------------------------------------------------------------------------
// Weight pre-pack: f32 row-major [K][N]  ->  bf16 packed [K/32][N][32].
// In the GEMM, lane n of a B fragment then reads two runs of 8 contiguous bf16
// (two 16-byte loads, no conversions) instead of 16 strided b32 + cvts.
// ---------------------------------------------------------------------------
__global__ void pack_w_k(const float* __restrict__ W, bf16* __restrict__ P,
                         int K, int N)
{
    int tid = blockIdx.x * blockDim.x + threadIdx.x;
    int total = (K >> 5) * N;
    if (tid >= total) return;
    int n  = tid % N;
    int kc = tid / N;
    const float* src = W + (size_t)kc * 32 * N + n;
    bf16* dst = P + ((size_t)kc * N + n) * 32;
#pragma unroll 8
    for (int j = 0; j < 32; ++j) dst[j] = (bf16)src[(size_t)j * N];
}

// ---------------------------------------------------------------------------
// Tiled GEMM:  C[rows, N] (+)= A_bf16[rows, K] @ Wp_bf16packed[K, N]
// Wave tile = 32 (M) x 64 (N): two A row-groups share every B fragment.
// Per 32-wide K step: 4 A loads + 8 B loads -> 8 v_wmma_f32_16x16x32_bf16.
// ---------------------------------------------------------------------------
__global__ __launch_bounds__(256) void gemm_wmma_bf16(
    const bf16* __restrict__ A, int lda,
    const bf16* __restrict__ Wp, int K, int N,
    float* __restrict__ Cf, bf16* __restrict__ Cb, int ldc,
    const float* __restrict__ G, int ldg,
    int row0, int Srows, int mtPB, int ntiles, int mode)
{
    int lane = threadIdx.x & 31;
    int wave = threadIdx.x >> 5;
    int gw   = blockIdx.x * 8 + wave;
    int total = B_ * mtPB * ntiles;
    if (gw >= total) return;

    int nt = gw % ntiles;
    int mt = gw / ntiles;
    int b  = mt / mtPB;
    int lm = (mt % mtPB) * 32;
    int n0 = nt * 64;
    int half = lane >> 4, l16 = lane & 15;

    int ar0 = lm + l16;        if (ar0 >= Srows) ar0 = Srows - 1;
    int ar1 = lm + 16 + l16;   if (ar1 >= Srows) ar1 = Srows - 1;
    const bf16* ap0 = A + (size_t)(b * ST + row0 + ar0) * lda;
    const bf16* ap1 = A + (size_t)(b * ST + row0 + ar1) * lda;

    v8f acc[2][4];
#pragma unroll
    for (int g = 0; g < 2; ++g)
#pragma unroll
        for (int t = 0; t < 4; ++t) acc[g][t] = zero8();

    for (int kc = 0; kc < K; kc += 32) {
        // A chunks (ISA 16-bit A 16x32 layout): lanes<16 K=kc+0..7 & +16..23
        int ka = kc + half * 8;
        v8bf a0lo = *(const v8bf*)(ap0 + ka);
        v8bf a0hi = *(const v8bf*)(ap0 + ka + 16);
        v8bf a1lo = *(const v8bf*)(ap1 + ka);
        v8bf a1hi = *(const v8bf*)(ap1 + ka + 16);
        v16bf a0, a1;
#pragma unroll
        for (int j = 0; j < 8; ++j) {
            a0[j] = a0lo[j]; a0[8 + j] = a0hi[j];
            a1[j] = a1lo[j]; a1[8 + j] = a1hi[j];
        }

#pragma unroll
        for (int t = 0; t < 4; ++t) {
            // packed B: lane's K-values = half*8+0..7 & +16..23 (matches A)
            const bf16* bp = Wp + ((size_t)(kc >> 5) * N + (n0 + t * 16 + l16)) * 32 + half * 8;
            v8bf blo = *(const v8bf*)bp;
            v8bf bhi = *(const v8bf*)(bp + 16);
            v16bf bv;
#pragma unroll
            for (int j = 0; j < 8; ++j) { bv[j] = blo[j]; bv[8 + j] = bhi[j]; }
            acc[0][t] = __builtin_amdgcn_wmma_f32_16x16x32_bf16(
                false, a0, false, bv, (short)0, acc[0][t], false, false);
            acc[1][t] = __builtin_amdgcn_wmma_f32_16x16x32_bf16(
                false, a1, false, bv, (short)0, acc[1][t], false, false);
        }
    }

#pragma unroll
    for (int g = 0; g < 2; ++g) {
#pragma unroll
        for (int t = 0; t < 4; ++t) {
#pragma unroll
            for (int r = 0; r < 8; ++r) {
                int row = lm + g * 16 + r + half * 8;
                if (row >= Srows) continue;
                size_t co = (size_t)(b * ST + row0 + row) * ldc + n0 + t * 16 + l16;
                float v = acc[g][t][r];
                if (mode == MODE_RESID) {
                    Cf[co] = Cf[co] + v;
                } else if (mode == MODE_GEGLU) {
                    size_t go = (size_t)(b * ST + row0 + row) * ldg + n0 + t * 16 + l16;
                    Cb[co] = (bf16)(gelu_tanh(G[go]) * v);
                } else {
                    Cf[co] = v;
                }
            }
        }
    }
}

// ---------------------------------------------------------------------------
// RMSNorm over D=1024, per token; weight chosen by mixture; bf16 output.
// ---------------------------------------------------------------------------
__global__ __launch_bounds__(256) void rmsnorm_k(
    const float* __restrict__ x, const float* __restrict__ wAll, int layer,
    bf16* __restrict__ out)
{
    int tok = blockIdx.x;                 // b*ST + s
    int s = tok % ST;
    int mix = (s < SV) ? 0 : ((s == SV) ? 1 : 2);
    const float* w  = wAll + (size_t)(mix * L_ + layer) * D_;
    const float* xr = x + (size_t)tok * D_;
    __shared__ float red[256];
    float p = 0.f;
    for (int d = threadIdx.x; d < D_; d += 256) { float v = xr[d]; p += v * v; }
    red[threadIdx.x] = p; __syncthreads();
    for (int st = 128; st > 0; st >>= 1) {
        if (threadIdx.x < st) red[threadIdx.x] += red[threadIdx.x + st];
        __syncthreads();
    }
    float inv = rsqrtf(red[0] / (float)D_ + 1e-6f);
    bf16* orow = out + (size_t)tok * D_;
    for (int d = threadIdx.x; d < D_; d += 256) orow[d] = (bf16)(xr[d] * inv * w[d]);
}

// ---------------------------------------------------------------------------
// RoPE on q (8 heads) & k (1 kv head) with mixture-local positions, + bf16
// conversion of q,k,v.
// ---------------------------------------------------------------------------
__global__ __launch_bounds__(128) void rope_convert_k(
    const float* __restrict__ qf, const float* __restrict__ kf,
    const float* __restrict__ vf,
    bf16* __restrict__ qb, bf16* __restrict__ kb, bf16* __restrict__ vb)
{
    int tok = blockIdx.x;
    int s = tok % ST;
    int ls = (s < SV) ? s : ((s == SV) ? 0 : (s - (SV + 1)));
    float fls = (float)ls;

    for (int idx = threadIdx.x; idx < NH * (HD / 2); idx += 128) {
        int h = idx >> 6, j = idx & 63;
        size_t base = (size_t)tok * QD + (size_t)h * HD;
        float invf = powf(10000.0f, -((float)j) / 64.0f);
        float ang = fls * invf, cs = cosf(ang), sn = sinf(ang);
        float x1 = qf[base + j], x2 = qf[base + 64 + j];
        qb[base + j]      = (bf16)(x1 * cs - x2 * sn);
        qb[base + 64 + j] = (bf16)(x2 * cs + x1 * sn);
    }
    if (threadIdx.x < 64) {
        int j = threadIdx.x;
        size_t base = (size_t)tok * KD;
        float invf = powf(10000.0f, -((float)j) / 64.0f);
        float ang = fls * invf, cs = cosf(ang), sn = sinf(ang);
        float x1 = kf[base + j], x2 = kf[base + 64 + j];
        kb[base + j]      = (bf16)(x1 * cs - x2 * sn);
        kb[base + 64 + j] = (bf16)(x2 * cs + x1 * sn);
    }
    {
        size_t base = (size_t)tok * KD;
        vb[base + threadIdx.x] = (bf16)vf[base + threadIdx.x];
    }
}

// ---------------------------------------------------------------------------
// Flash attention, one wave = one (b, h, 16-row q-tile).  32-key tiles:
// S = Q·K^T via 2x4 chained bf16 WMMAs; tanh softcap + mask + streaming
// softmax in the WMMA C layout (row reductions = 16-lane shfl_xor);
// P transposed through LDS into A layout; P·V via 8 WMMAs.
// ---------------------------------------------------------------------------
__global__ __launch_bounds__(128) void flash_attn_k(
    const bf16* __restrict__ qb, const bf16* __restrict__ kb,
    const bf16* __restrict__ vb, const float* __restrict__ mask,
    bf16* __restrict__ ao)
{
    __shared__ __bf16 Vs[4][32 * HD];   // per-wave V tile (32 keys x 128)
    __shared__ __bf16 Ps[4][16 * 32];   // per-wave P tile (16 q x 32 keys)

    int lane = threadIdx.x & 31, wave = threadIdx.x >> 5;
    int gw = blockIdx.x * 4 + wave;
    int qt = gw % QTILES;
    int bh = gw / QTILES;
    int h = bh % NH, b = bh / NH;
    int half = lane >> 4, l16 = lane & 15;
    __bf16* vs = Vs[wave];
    __bf16* ps = Ps[wave];

    // Resident Q tile: 4 K-chunks in A layout
    int qrow = qt * 16 + l16;
    int qrc  = qrow < ST ? qrow : ST - 1;
    const bf16* qp = qb + (size_t)(b * ST + qrc) * QD + (size_t)h * HD;
    v16bf aq[4];
#pragma unroll
    for (int c = 0; c < 4; ++c) {
        int ka = c * 32 + half * 16;
        v8bf lo = *(const v8bf*)(qp + ka);
        v8bf hi = *(const v8bf*)(qp + ka + 8);
#pragma unroll
        for (int j = 0; j < 8; ++j) { aq[c][j] = lo[j]; aq[c][8 + j] = hi[j]; }
    }

    v8f O[8];
    float mrow[8], lrow[8];
#pragma unroll
    for (int t = 0; t < 8; ++t) { O[t] = zero8(); mrow[t] = -3.0e38f; lrow[t] = 0.f; }

    for (int kt = 0; kt < 19; ++kt) {
        int k0 = kt * 32;

        // Stage V tile (bf16) into LDS: lane -> key k0+lane, 16B chunks
        {
            int key = k0 + lane;
            if (key < ST) {
                const bf16* vp = vb + (size_t)(b * ST + key) * KD;
#pragma unroll
                for (int j = 0; j < HD / 8; ++j)
                    *(v8bf*)(vs + lane * HD + j * 8) = *(const v8bf*)(vp + j * 8);
            } else {
                v8bf z;
#pragma unroll
                for (int e = 0; e < 8; ++e) z[e] = (bf16)0.0f;
#pragma unroll
                for (int j = 0; j < HD / 8; ++j)
                    *(v8bf*)(vs + lane * HD + j * 8) = z;
            }
        }

        // Logits: S0 = keys k0..k0+15, S1 = keys k0+16..k0+31
        v8f s0 = zero8(), s1 = zero8();
#pragma unroll
        for (int c = 0; c < 4; ++c) {
            int key0 = k0 + l16;       int k0c = key0 < ST ? key0 : ST - 1;
            int key1 = k0 + 16 + l16;  int k1c = key1 < ST ? key1 : ST - 1;
            const bf16* kp0 = kb + (size_t)(b * ST + k0c) * KD + c * 32 + half * 16;
            const bf16* kp1 = kb + (size_t)(b * ST + k1c) * KD + c * 32 + half * 16;
            v8bf a0 = *(const v8bf*)kp0, b0 = *(const v8bf*)(kp0 + 8);
            v8bf a1 = *(const v8bf*)kp1, b1 = *(const v8bf*)(kp1 + 8);
            v16bf bv0, bv1;
#pragma unroll
            for (int j = 0; j < 8; ++j) {
                bv0[j] = a0[j]; bv0[8 + j] = b0[j];
                bv1[j] = a1[j]; bv1[8 + j] = b1[j];
            }
            s0 = __builtin_amdgcn_wmma_f32_16x16x32_bf16(false, aq[c], false, bv0, (short)0, s0, false, false);
            s1 = __builtin_amdgcn_wmma_f32_16x16x32_bf16(false, aq[c], false, bv1, (short)0, s1, false, false);
        }

        // Softcap + mask + streaming softmax (row = r + 8*half per lane-half)
        int key0 = k0 + l16, key1 = k0 + 16 + l16;
#pragma unroll
        for (int r = 0; r < 8; ++r) {
            int qr = qt * 16 + r + half * 8;
            float x0 = tanhf(s0[r] * INV_SQRT_HD * (1.0f / SOFTCAP)) * SOFTCAP;
            float x1 = tanhf(s1[r] * INV_SQRT_HD * (1.0f / SOFTCAP)) * SOFTCAP;
            if (qr < ST) {
                const float* mr = mask + ((size_t)b * ST + qr) * ST;
                if (key0 < ST) x0 += mr[key0];
                if (key1 < ST) x1 += mr[key1];
            }
            if (key0 >= ST) x0 = -1e30f;
            if (key1 >= ST) x1 = -1e30f;

            float rm = fmaxf(x0, x1);
            rm = fmaxf(rm, __shfl_xor(rm, 1, 32));
            rm = fmaxf(rm, __shfl_xor(rm, 2, 32));
            rm = fmaxf(rm, __shfl_xor(rm, 4, 32));
            rm = fmaxf(rm, __shfl_xor(rm, 8, 32));
            float mn = fmaxf(mrow[r], rm);
            float sc = __expf(mrow[r] - mn);
            float p0 = __expf(x0 - mn), p1 = __expf(x1 - mn);
            float psum = p0 + p1;
            psum += __shfl_xor(psum, 1, 32);
            psum += __shfl_xor(psum, 2, 32);
            psum += __shfl_xor(psum, 4, 32);
            psum += __shfl_xor(psum, 8, 32);
            lrow[r] = lrow[r] * sc + psum;
            mrow[r] = mn;
#pragma unroll
            for (int t = 0; t < 8; ++t) O[t][r] *= sc;

            ps[(r + half * 8) * 32 + l16]      = (bf16)p0;
            ps[(r + half * 8) * 32 + 16 + l16] = (bf16)p1;
        }
        __syncthreads();

        // Read P back in A layout: row = l16; K = half*8+0..7 & +16..23
        v16bf ap;
        {
            const __bf16* pr = ps + l16 * 32 + half * 8;
#pragma unroll
            for (int j = 0; j < 8; ++j) { ap[j] = pr[j]; ap[8 + j] = pr[16 + j]; }
        }
        // P·V: 8 hd-subtiles of 16 columns
#pragma unroll
        for (int t = 0; t < 8; ++t) {
            v16bf bv;
#pragma unroll
            for (int j = 0; j < 16; ++j)
                bv[j] = vs[(half * 8 + j + (j >> 3) * 8) * HD + t * 16 + l16];
            O[t] = __builtin_amdgcn_wmma_f32_16x16x32_bf16(false, ap, false, bv, (short)0, O[t], false, false);
        }
        __syncthreads();
    }

    // Normalize and store [b, s, h*HD + col] as bf16
#pragma unroll
    for (int t = 0; t < 8; ++t) {
#pragma unroll
        for (int r = 0; r < 8; ++r) {
            int row = qt * 16 + r + half * 8;
            if (row < ST) {
                float v = O[t][r] / lrow[r];
                ao[(size_t)(b * ST + row) * QD + (size_t)h * HD + t * 16 + l16] = (bf16)v;
            }
        }
    }
}

// ---------------------------------------------------------------------------
__global__ void init_emb_k(const float* __restrict__ ev, const float* __restrict__ ep,
                           const float* __restrict__ ea, float* __restrict__ emb)
{
    size_t i = (size_t)blockIdx.x * blockDim.x + threadIdx.x;
    size_t n = (size_t)B_ * ST * D_;
    if (i >= n) return;
    int d = (int)(i % D_);
    size_t tok = i / D_;
    int s = (int)(tok % ST);
    int b = (int)(tok / ST);
    float v;
    if (s < SV)       v = ev[((size_t)b * SV + s) * D_ + d];
    else if (s == SV) v = ep[(size_t)b * D_ + d];
    else              v = ea[((size_t)b * SA + (s - SV - 1)) * D_ + d];
    emb[i] = v * 32.0f;   // sqrt(1024)
}

__global__ void final_copy_k(const float* __restrict__ emb, float* __restrict__ out)
{
    size_t i = (size_t)blockIdx.x * blockDim.x + threadIdx.x;
    size_t n = (size_t)B_ * SA * D_;
    if (i >= n) return;
    int d = (int)(i % D_);
    size_t t = i / D_;
    int a = (int)(t % SA);
    int b = (int)(t / SA);
    out[i] = emb[((size_t)b * ST + SV + 1 + a) * D_ + d];
}

// ---------------------------------------------------------------------------
extern "C" void kernel_launch(void* const* d_in, const int* in_sizes, int n_in,
                              void* d_out, int out_size, void* d_ws, size_t ws_size,
                              hipStream_t stream)
{
    (void)in_sizes; (void)n_in; (void)out_size; (void)ws_size;
    const float* mask = (const float*)d_in[0];
    const float* ev   = (const float*)d_in[1];
    const float* ep   = (const float*)d_in[2];
    const float* ea   = (const float*)d_in[3];
    const float* Wq   = (const float*)d_in[4];
    const float* Wk   = (const float*)d_in[5];
    const float* Wv   = (const float*)d_in[6];
    const float* Wo   = (const float*)d_in[7];
    const float* n1   = (const float*)d_in[8];
    const float* n2   = (const float*)d_in[9];
    const float* Wg   = (const float*)d_in[10];
    const float* Wu   = (const float*)d_in[11];
    const float* Wd   = (const float*)d_in[12];
    // d_in[13..15] = pos arrays (arange); RoPE uses mixture-local indices.

    char* ws = (char*)d_ws;
    size_t off = 0;
    auto carve = [&](size_t bytes) -> void* {
        void* p = ws + off;
        off += (bytes + 255) & ~(size_t)255;
        return p;
    };
    float* emb  = (float*)carve((size_t)B_ * ST * D_ * 4);
    bf16*  hnb  = (bf16*) carve((size_t)B_ * ST * D_ * 2);
    float* qf   = (float*)carve((size_t)B_ * ST * QD * 4);
    float* kf   = (float*)carve((size_t)B_ * ST * KD * 4);
    float* vf   = (float*)carve((size_t)B_ * ST * KD * 4);
    bf16*  qbb  = (bf16*) carve((size_t)B_ * ST * QD * 2);
    bf16*  kbb  = (bf16*) carve((size_t)B_ * ST * KD * 2);
    bf16*  vbb  = (bf16*) carve((size_t)B_ * ST * KD * 2);
    bf16*  aob  = (bf16*) carve((size_t)B_ * ST * QD * 2);
    float* gbuf = (float*)carve((size_t)B_ * ST * DFF * 4);
    bf16*  actb = (bf16*) carve((size_t)B_ * ST * DFF * 2);
    // per-layer packed weights (bf16, reused across layers)
    bf16* pWq = (bf16*)carve((size_t)3 * D_  * QD  * 2);
    bf16* pWk = (bf16*)carve((size_t)3 * D_  * KD  * 2);
    bf16* pWv = (bf16*)carve((size_t)3 * D_  * KD  * 2);
    bf16* pWo = (bf16*)carve((size_t)3 * QD  * D_  * 2);
    bf16* pWg = (bf16*)carve((size_t)3 * D_  * DFF * 2);
    bf16* pWu = (bf16*)carve((size_t)3 * D_  * DFF * 2);
    bf16* pWd = (bf16*)carve((size_t)3 * DFF * D_  * 2);

    const int mixStart[3] = {0, SV, SV + 1};
    const int mixLen[3]   = {SV, SP, SA};

    {
        size_t n = (size_t)B_ * ST * D_;
        init_emb_k<<<(unsigned)((n + 255) / 256), 256, 0, stream>>>(ev, ep, ea, emb);
    }

    auto packw = [&](const float* src, bf16* dst, int K, int N) {
        int total = (K >> 5) * N;
        pack_w_k<<<(total + 255) / 256, 256, 0, stream>>>(src, dst, K, N);
    };

    auto gemm = [&](const bf16* A, int lda, const bf16* Wp, int K, int N,
                    float* Cf, bf16* Cb, int ldc, const float* G, int ldg,
                    int mix, int mode) {
        int mtPB = (mixLen[mix] + 31) / 32;
        int ntiles = N / 64;
        int total = B_ * mtPB * ntiles;
        int blocks = (total + 7) / 8;
        gemm_wmma_bf16<<<blocks, 256, 0, stream>>>(A, lda, Wp, K, N, Cf, Cb, ldc, G, ldg,
                                                   mixStart[mix], mixLen[mix], mtPB, ntiles, mode);
    };

    for (int l = 0; l < L_; ++l) {
        // Pack this layer's weights into WMMA-B bf16 layout
        for (int mix = 0; mix < 3; ++mix) {
            packw(Wq + (size_t)(mix * L_ + l) * D_ * QD,   pWq + (size_t)mix * D_ * QD,   D_,  QD);
            packw(Wk + (size_t)(mix * L_ + l) * D_ * KD,   pWk + (size_t)mix * D_ * KD,   D_,  KD);
            packw(Wv + (size_t)(mix * L_ + l) * D_ * KD,   pWv + (size_t)mix * D_ * KD,   D_,  KD);
            packw(Wo + (size_t)(mix * L_ + l) * QD * D_,   pWo + (size_t)mix * QD * D_,   QD,  D_);
            packw(Wg + (size_t)(mix * L_ + l) * D_ * DFF,  pWg + (size_t)mix * D_ * DFF,  D_,  DFF);
            packw(Wu + (size_t)(mix * L_ + l) * D_ * DFF,  pWu + (size_t)mix * D_ * DFF,  D_,  DFF);
            packw(Wd + (size_t)(mix * L_ + l) * DFF * D_,  pWd + (size_t)mix * DFF * D_,  DFF, D_);
        }

        rmsnorm_k<<<B_ * ST, 256, 0, stream>>>(emb, n1, l, hnb);
        for (int mix = 0; mix < 3; ++mix) {
            gemm(hnb, D_, pWq + (size_t)mix * D_ * QD, D_, QD, qf, nullptr, QD, nullptr, 0, mix, MODE_PLAIN);
            gemm(hnb, D_, pWk + (size_t)mix * D_ * KD, D_, KD, kf, nullptr, KD, nullptr, 0, mix, MODE_PLAIN);
            gemm(hnb, D_, pWv + (size_t)mix * D_ * KD, D_, KD, vf, nullptr, KD, nullptr, 0, mix, MODE_PLAIN);
        }
        rope_convert_k<<<B_ * ST, 128, 0, stream>>>(qf, kf, vf, qbb, kbb, vbb);
        flash_attn_k<<<(B_ * NH * QTILES) / 4, 128, 0, stream>>>(qbb, kbb, vbb, mask, aob);

        int mlo = (l == L_ - 1) ? 2 : 0;   // last layer skips post-attn for vlm & proprio
        for (int mix = mlo; mix < 3; ++mix)
            gemm(aob, QD, pWo + (size_t)mix * QD * D_, QD, D_, emb, nullptr, D_, nullptr, 0, mix, MODE_RESID);
        rmsnorm_k<<<B_ * ST, 256, 0, stream>>>(emb, n2, l, hnb);
        for (int mix = mlo; mix < 3; ++mix) {
            gemm(hnb, D_, pWg + (size_t)mix * D_ * DFF, D_, DFF, gbuf, nullptr, DFF, nullptr, 0, mix, MODE_PLAIN);
            gemm(hnb, D_, pWu + (size_t)mix * D_ * DFF, D_, DFF, nullptr, actb, DFF, gbuf, DFF, mix, MODE_GEGLU);
            gemm(actb, DFF, pWd + (size_t)mix * DFF * D_, DFF, D_, emb, nullptr, D_, nullptr, 0, mix, MODE_RESID);
        }
    }

    {
        size_t n = (size_t)B_ * SA * D_;
        final_copy_k<<<(unsigned)((n + 255) / 256), 256, 0, stream>>>(emb, (float*)d_out);
    }
}
